// ConvBranch_38259568673464
// MI455X (gfx1250) — matrix operand order repaired and checked by reading, not compiled, not run on hardware
//
#include <hip/hip_runtime.h>
#include <hip/hip_bf16.h>

// ---------------- problem constants (match reference) ----------------
#define B_   4
#define T_   2048
#define F_   128
#define DM_  512
#define L_   4
#define STRIDE_ 4
#define KK_  8            // front conv kernel
#define N_   16           // SSM state
#define DC_  4            // depthwise conv taps
#define DI_  1024         // inner dim
#define R_   32           // dt rank
#define TS_  (T_/STRIDE_) // 512
#define MROWS (B_*TS_)    // 2048 rows for all GEMMs
#define EPS_ 1e-5f

typedef __attribute__((ext_vector_type(16))) _Float16 v16h;
typedef __attribute__((ext_vector_type(8)))  float    v8f;

__device__ __forceinline__ float sigmoidf_(float x){ return 1.0f/(1.0f+expf(-x)); }
__device__ __forceinline__ float siluf_(float x){ return x*sigmoidf_(x); }
__device__ __forceinline__ float geluf_(float x){ return 0.5f*x*(1.0f+erff(x*0.70710678118654752f)); }
__device__ __forceinline__ float softplusf_(float x){ return (x>20.0f)? x : log1pf(expf(x)); }

// pack 8 f32 (two float4) into halves v[o..o+7]
__device__ __forceinline__ void cvt8_(v16h& v, int o, const float4& p, const float4& q){
  v[o+0]=(_Float16)p.x; v[o+1]=(_Float16)p.y; v[o+2]=(_Float16)p.z; v[o+3]=(_Float16)p.w;
  v[o+4]=(_Float16)q.x; v[o+5]=(_Float16)q.y; v[o+6]=(_Float16)q.z; v[o+7]=(_Float16)q.w;
}

// ---------------- front strided causal conv + GELU ----------------
__global__ void front_conv_gelu_k(const float* __restrict__ x,
                                  const float* __restrict__ cw,
                                  const float* __restrict__ cb,
                                  float* __restrict__ h)
{
  int idx = blockIdx.x*blockDim.x + threadIdx.x;
  if (idx >= MROWS*DM_) return;
  int d   = idx % DM_;
  int row = idx / DM_;
  int t   = row % TS_;
  int b   = row / TS_;
  float s = cb[d];
  #pragma unroll
  for (int k = 0; k < KK_; ++k) {
    int ti = t*STRIDE_ + k - (KK_-1);
    if (ti < 0) continue;
    const float* xr = x  + (size_t)(b*T_+ti)*F_;
    const float* wr = cw + (size_t)d*F_*KK_ + k;
    float acc = 0.f;
    for (int f = 0; f < F_; ++f) acc = fmaf(xr[f], wr[(size_t)f*KK_], acc);
    s += acc;
  }
  h[idx] = geluf_(s);
}

// ---------------- LayerNorm over DM=512, one block per row ----------------
__global__ void layernorm_k(const float* __restrict__ in, float* __restrict__ out,
                            const float* __restrict__ w, const float* __restrict__ bb)
{
  __shared__ float red[256];
  int row = blockIdx.x;
  int tid = threadIdx.x;
  const float* xr = in + (size_t)row*DM_;
  float a0 = xr[tid], a1 = xr[tid+256];
  red[tid] = a0 + a1;
  __syncthreads();
  for (int s = 128; s > 0; s >>= 1) { if (tid < s) red[tid] += red[tid+s]; __syncthreads(); }
  float mean = red[0] * (1.0f/DM_);
  __syncthreads();
  float d0 = a0-mean, d1 = a1-mean;
  red[tid] = d0*d0 + d1*d1;
  __syncthreads();
  for (int s = 128; s > 0; s >>= 1) { if (tid < s) red[tid] += red[tid+s]; __syncthreads(); }
  float inv = rsqrtf(red[0]*(1.0f/DM_) + EPS_);
  float* orow = out + (size_t)row*DM_;
  orow[tid]     = d0*inv*w[tid]     + bb[tid];
  orow[tid+256] = d1*inv*w[tid+256] + bb[tid+256];
}

// ---------------- WMMA GEMM: C[m,n] = sum_k A[m,k]*W[n,k] ----------------
// A: MxK row-major (row stride lda), W: NxK row-major, C: MxN row-major.
// Block = (32,4): 64x64 macro-tile; each wave owns a 16x64 strip.
// The shared 64x32 B panel is staged into LDS once per K-step with
// GLOBAL_LOAD_ASYNC_TO_LDS_B128 (ASYNCcnt), then consumed by all 4 waves;
// A fragments come straight from global (b128) while the DMA is in flight.
// MODE 0: store; MODE 1: softplus(v + bias[n]) store; MODE 2: C += v (residual).
template<int MODE>
__global__ __launch_bounds__(128)
void gemm_wmma_k(const float* __restrict__ A, const float* __restrict__ W,
                 float* __restrict__ C, const float* __restrict__ bias,
                 int Kd, int Nd, int lda)
{
  __shared__ float bsm[64*32];           // 8 KB B panel: row-major 64 rows x 32 floats
  const int lane = threadIdx.x;          // 0..31 (one wave per threadIdx.y)
  const int wv   = threadIdx.y;
  const int tid  = wv*32 + lane;         // 0..127
  const int half16 = lane >> 4;
  const int l15    = lane & 15;
  const int m0 = blockIdx.y*64 + wv*16;
  const int n0 = blockIdx.x*64;

  const float* arow = A + (size_t)(m0 + l15)*lda;
  // low 32 bits of a generic pointer to __shared__ == workgroup-relative LDS addr
  const unsigned bsm_base = (unsigned)(size_t)&bsm[0];

  v8f acc0 = {}, acc1 = {}, acc2 = {}, acc3 = {};

  const int abase = half16 ? 8 : 0;      // A: lanes<16 take K {0..7,16..23}; lanes>=16 {8..15,24..31}
  const int bbase = half16 ? 8 : 0;      // B: lanes<16 take K {0..7,16..23}; lanes>=16 {8..15,24..31}

  for (int k0 = 0; k0 < Kd; k0 += 32) {
    // async-DMA the 64x32 B panel: 512 x 16B chunks, 4 per thread
    #pragma unroll
    for (int i = 0; i < 4; ++i) {
      int c    = tid + 128*i;            // chunk id
      int rowq = c >> 3;                 // panel row 0..63
      int q    = c & 7;                  // 16B chunk within row
      unsigned long long ga =
          (unsigned long long)(W + (size_t)(n0 + rowq)*Kd + k0 + q*4);
      unsigned ldst = bsm_base + (unsigned)(rowq*128 + q*16);
      asm volatile("global_load_async_to_lds_b128 %0, %1, off"
                   :: "v"(ldst), "v"(ga) : "memory");
    }

    // A fragment from global while the DMA is in flight
    const float4* ap0 = reinterpret_cast<const float4*>(arow + k0 + abase);
    const float4* ap1 = reinterpret_cast<const float4*>(arow + k0 + abase + 16);
    v16h av;
    cvt8_(av, 0, ap0[0], ap0[1]);
    cvt8_(av, 8, ap1[0], ap1[1]);

    asm volatile("s_wait_asynccnt 0" ::: "memory");  // my wave's DMA done
    __syncthreads();                                  // everyone's DMA visible

    // B fragments from LDS (ds_load_b128)
    v16h bv0, bv1, bv2, bv3;
    {
      const float4* bp = reinterpret_cast<const float4*>(&bsm[( 0 + l15)*32 + bbase]);
      cvt8_(bv0, 0, bp[0], bp[1]); cvt8_(bv0, 8, bp[4], bp[5]);
    }
    {
      const float4* bp = reinterpret_cast<const float4*>(&bsm[(16 + l15)*32 + bbase]);
      cvt8_(bv1, 0, bp[0], bp[1]); cvt8_(bv1, 8, bp[4], bp[5]);
    }
    {
      const float4* bp = reinterpret_cast<const float4*>(&bsm[(32 + l15)*32 + bbase]);
      cvt8_(bv2, 0, bp[0], bp[1]); cvt8_(bv2, 8, bp[4], bp[5]);
    }
    {
      const float4* bp = reinterpret_cast<const float4*>(&bsm[(48 + l15)*32 + bbase]);
      cvt8_(bv3, 0, bp[0], bp[1]); cvt8_(bv3, 8, bp[4], bp[5]);
    }

    acc0 = __builtin_amdgcn_wmma_f32_16x16x32_f16(false, av, false, bv0, (short)0, acc0, false, false);
    acc1 = __builtin_amdgcn_wmma_f32_16x16x32_f16(false, av, false, bv1, (short)0, acc1, false, false);
    acc2 = __builtin_amdgcn_wmma_f32_16x16x32_f16(false, av, false, bv2, (short)0, acc2, false, false);
    acc3 = __builtin_amdgcn_wmma_f32_16x16x32_f16(false, av, false, bv3, (short)0, acc3, false, false);

    __syncthreads();                     // panel consumed; safe to overwrite
  }

  // D layout: VGPR r -> lanes 0-15: (M=m0+r, N=n0+lane); lanes 16-31: (M=m0+8+r)
  #pragma unroll
  for (int j = 0; j < 4; ++j) {
    const v8f& acc = (j==0)?acc0:(j==1)?acc1:(j==2)?acc2:acc3;
    int col = n0 + j*16 + l15;
    float bcol = (MODE == 1) ? bias[col] : 0.0f;
    #pragma unroll
    for (int r = 0; r < 8; ++r) {
      int rowi = m0 + r + half16*8;
      size_t o = (size_t)rowi*Nd + col;
      float v = acc[r];
      if (MODE == 0)      C[o] = v;
      else if (MODE == 1) C[o] = softplusf_(v + bcol);
      else                C[o] += v;
    }
  }
}

// ---------------- depthwise causal conv (DC=4) + SiLU ----------------
__global__ void dwconv_silu_k(const float* __restrict__ xz, const float* __restrict__ cw,
                              const float* __restrict__ cb, float* __restrict__ xi2)
{
  int idx = blockIdx.x*blockDim.x + threadIdx.x;
  if (idx >= MROWS*DI_) return;
  int d   = idx % DI_;
  int row = idx / DI_;
  int t   = row % TS_;
  int b   = row / TS_;
  float s = cb[d];
  #pragma unroll
  for (int j = 0; j < DC_; ++j) {
    int tt = t - (DC_-1) + j;
    if (tt >= 0)
      s = fmaf(xz[(size_t)(b*TS_+tt)*(2*DI_) + d], cw[d*DC_ + j], s);
  }
  xi2[idx] = siluf_(s);
}

// ---------------- selective scan: one thread per (b,d), state N=16 in regs ----------------
__global__ void scan_k(const float* __restrict__ dt, const float* __restrict__ u,
                       const float* __restrict__ xdbl, const float* __restrict__ A_log,
                       float* __restrict__ y)
{
  int idx = blockIdx.x*blockDim.x + threadIdx.x;
  if (idx >= B_*DI_) return;
  int d = idx % DI_;
  int b = idx / DI_;
  float A[N_];
  #pragma unroll
  for (int n = 0; n < N_; ++n) A[n] = -expf(A_log[(size_t)d*N_ + n]);
  float hs[N_];
  #pragma unroll
  for (int n = 0; n < N_; ++n) hs[n] = 0.f;
  for (int t = 0; t < TS_; ++t) {
    size_t rb = (size_t)(b*TS_ + t);
    const float* xr = xdbl + rb*64;              // [dt(32) | B(16) | C(16)]
    if (t + 1 < TS_) __builtin_prefetch(xdbl + (rb+1)*64, 0, 1);  // global_prefetch_b8
    float dtv = dt[rb*DI_ + d];
    float uv  = u [rb*DI_ + d];
    float du  = dtv*uv;
    float yv  = 0.f;
    #pragma unroll
    for (int n = 0; n < N_; ++n) {
      float dA = expf(dtv * A[n]);
      hs[n] = fmaf(dA, hs[n], du * xr[R_ + n]);         // h = dA*h + dt*u*B
      yv    = fmaf(hs[n], xr[R_ + N_ + n], yv);         // y += h*C
    }
    y[rb*DI_ + d] = yv;
  }
}

// ---------------- gate: y2 = (y + xi2*D) * silu(z) ----------------
__global__ void gate_k(const float* __restrict__ y, const float* __restrict__ xi2,
                       const float* __restrict__ xz, const float* __restrict__ Dp,
                       float* __restrict__ y2)
{
  int idx = blockIdx.x*blockDim.x + threadIdx.x;
  if (idx >= MROWS*DI_) return;
  int d   = idx % DI_;
  int row = idx / DI_;
  float z = xz[(size_t)row*(2*DI_) + DI_ + d];
  y2[idx] = fmaf(xi2[idx], Dp[d], y[idx]) * siluf_(z);
}

// ---------------- repeat_interleave x4 upsample ----------------
__global__ void upsample_k(const float* __restrict__ hn, float* __restrict__ out)
{
  int idx = blockIdx.x*blockDim.x + threadIdx.x;
  if (idx >= B_*T_*DM_) return;
  int d = idx % DM_;
  int t = (idx / DM_) % T_;
  int b = idx / (DM_*T_);
  out[idx] = hn[(size_t)(b*TS_ + (t >> 2))*DM_ + d];
}

extern "C" void kernel_launch(void* const* d_in, const int* in_sizes, int n_in,
                              void* d_out, int out_size, void* d_ws, size_t ws_size,
                              hipStream_t stream)
{
  (void)in_sizes; (void)n_in; (void)out_size; (void)ws_size;

  const float* x          = (const float*)d_in[0];
  const float* conv_w     = (const float*)d_in[1];
  const float* conv_b     = (const float*)d_in[2];
  const float* in_proj_w  = (const float*)d_in[3];
  const float* conv1d_w   = (const float*)d_in[4];
  const float* conv1d_b   = (const float*)d_in[5];
  const float* x_proj_w   = (const float*)d_in[6];
  const float* dt_proj_w  = (const float*)d_in[7];
  const float* dt_proj_b  = (const float*)d_in[8];
  const float* A_log      = (const float*)d_in[9];
  const float* D_skip     = (const float*)d_in[10];
  const float* ln_w       = (const float*)d_in[11];
  const float* ln_b       = (const float*)d_in[12];
  const float* out_proj_w = (const float*)d_in[13];
  const float* norm_w     = (const float*)d_in[14];
  const float* norm_b     = (const float*)d_in[15];
  float* out = (float*)d_out;

  // workspace layout (f32)
  float* ws = (float*)d_ws;
  size_t off = 0;
  float* h    = ws + off; off += (size_t)MROWS*DM_;      // residual stream
  float* hn   = ws + off; off += (size_t)MROWS*DM_;      // normalized
  float* xz   = ws + off; off += (size_t)MROWS*2*DI_;    // in_proj output
  float* xi2  = ws + off; off += (size_t)MROWS*DI_;      // conv+silu
  float* xdbl = ws + off; off += (size_t)MROWS*64;       // x_proj output
  float* dtb  = ws + off; off += (size_t)MROWS*DI_;      // softplus(dt)
  float* yb   = ws + off; off += (size_t)MROWS*DI_;      // scan output
  float* y2   = ws + off; off += (size_t)MROWS*DI_;      // gated

  front_conv_gelu_k<<<(MROWS*DM_+255)/256, 256, 0, stream>>>(x, conv_w, conv_b, h);

  for (int l = 0; l < L_; ++l) {
    layernorm_k<<<MROWS, 256, 0, stream>>>(h, hn, ln_w + l*DM_, ln_b + l*DM_);

    // xz = hn @ in_proj_w^T  (2048 x 512 x 2048)
    gemm_wmma_k<0><<<dim3((2*DI_)/64, MROWS/64), dim3(32,4), 0, stream>>>(
        hn, in_proj_w + (size_t)l*2*DI_*DM_, xz, nullptr, DM_, 2*DI_, DM_);

    dwconv_silu_k<<<(MROWS*DI_+255)/256, 256, 0, stream>>>(
        xz, conv1d_w + l*DI_*DC_, conv1d_b + l*DI_, xi2);

    // xdbl = xi2 @ x_proj_w^T  (2048 x 1024 x 64)
    gemm_wmma_k<0><<<dim3(64/64, MROWS/64), dim3(32,4), 0, stream>>>(
        xi2, x_proj_w + (size_t)l*64*DI_, xdbl, nullptr, DI_, 64, DI_);

    // dt = softplus(xdbl[:, :R] @ dt_proj_w^T + b)  (2048 x 32 x 1024, single WMMA step)
    gemm_wmma_k<1><<<dim3(DI_/64, MROWS/64), dim3(32,4), 0, stream>>>(
        xdbl, dt_proj_w + (size_t)l*DI_*R_, dtb, dt_proj_b + l*DI_, R_, DI_, 64);

    scan_k<<<(B_*DI_+255)/256, 256, 0, stream>>>(
        dtb, xi2, xdbl, A_log + (size_t)l*DI_*N_, yb);

    gate_k<<<(MROWS*DI_+255)/256, 256, 0, stream>>>(yb, xi2, xz, D_skip + l*DI_, y2);

    // h += y2 @ out_proj_w^T  (2048 x 1024 x 512, accumulate epilogue = residual)
    gemm_wmma_k<2><<<dim3(DM_/64, MROWS/64), dim3(32,4), 0, stream>>>(
        y2, out_proj_w + (size_t)l*DM_*DI_, h, nullptr, DI_, DM_, DI_);
  }

  layernorm_k<<<MROWS, 256, 0, stream>>>(h, hn, norm_w, norm_b);
  upsample_k<<<(B_*T_*DM_+255)/256, 256, 0, stream>>>(hn, out);
}
